// LegacyURNN_28235115004146
// MI455X (gfx1250) — matrix operand order repaired and checked by reading, not compiled, not run on hardware
//
#include <hip/hip_runtime.h>
#include <math.h>

// ---------------------------------------------------------------------------
// LegacyURNN fused kernel for MI455X (gfx1250, wave32, WMMA)
// B=16384, H=512, I=256
// ---------------------------------------------------------------------------

typedef __bf16 bf16_t;
typedef __attribute__((ext_vector_type(16))) __bf16 v16bf;
typedef __attribute__((ext_vector_type(8)))  float  v8f;
typedef __attribute__((ext_vector_type(4)))  int    v4i;

// Async global->LDS staging (gfx1250): hide the hx fetch behind the WMMA GEMM.
#if defined(__AMDGCN__) && \
    __has_builtin(__builtin_amdgcn_global_load_async_to_lds_b128) && \
    __has_builtin(__builtin_amdgcn_s_wait_asynccnt)
#define URNN_ASYNC_LDS 1
typedef __attribute__((address_space(1))) v4i gv4i;   // global-AS int4
typedef __attribute__((address_space(3))) v4i lv4i;   // LDS-AS int4
#else
#define URNN_ASYNC_LDS 0
#endif

#define B_ROWS 16384
#define HDIM   512
#define IDIM   256
#define ROWS   16          // batch rows per block
#define INV_N  (1.0f / 512.0f)

// parameter block layout (floats) inside workspace
#define PD1C 0
#define PD1S 512
#define PD2C 1024
#define PD2S 1536
#define PD3C 2048
#define PD3S 2560
#define PV1R 3072
#define PV1I 3584
#define PV2R 4096
#define PV2I 4608
#define NPAR 5120
#define PBETA 5120          // beta appended in LDS pbuf (not in ws par block)

// LDS layout in 32-bit words
#define L_HRE   0
#define L_HIM   8192
#define L_TRE   16384
#define L_TIM   24576
#define L_ERE   32768
#define L_EIM   40960
#define L_RED   49152       // 512 (dot-product reduction scratch)
#define L_PBUF  49664       // 5120 params + 512 beta
#define L_PERM  55296       // 512 ints
#define L_WORDS 55808
#define SMEM_BYTES (L_WORDS * 4)

union Frag { v16bf v; struct { uint4 lo, hi; } q; };

#define WMMA_BF16(A, Bv, C) \
  __builtin_amdgcn_wmma_f32_16x16x32_bf16(false, (A), false, (Bv), (short)0, (C), false, false)

// A fragment: 16-bit A 16x32 in contiguous-K order (matches load_fragB).
// lane L: M = L&15; kb=(L>>4)*16 -> elems 0..15 = K kb..kb+15 (contiguous).
__device__ __forceinline__ v16bf load_fragA(const bf16_t* __restrict__ base,
                                            int m, int k0, int kb) {
  Frag f;
  const bf16_t* p = base + m * IDIM + k0 + kb;
  f.q.lo = *(const uint4*)p;
  f.q.hi = *(const uint4*)(p + 8);
  return f.v;
}

// B fragment: lane n (n=L&15) holds column N=n; lanes 0-15 hold K=0..15,
// lanes 16-31 hold K=16..31 (ks=(L>>4)*16) -> 16 contiguous K values.
__device__ __forceinline__ v16bf load_fragB(const bf16_t* __restrict__ W,
                                            int h, int k0, int ks) {
  Frag f;
  const bf16_t* p = W + h * IDIM + k0 + ks;
  f.q.lo = *(const uint4*)p;
  f.q.hi = *(const uint4*)(p + 8);
  return f.v;
}

// 16 rows x 512-point radix-2 DIT FFT (input in bit-reversed order).
// sign = -1 forward, +1 inverse (unnormalized).
__device__ __forceinline__ void fft16x512(float* __restrict__ re,
                                          float* __restrict__ im,
                                          float sign, int tid) {
  for (int len = 2; len <= 512; len <<= 1) {
    const int half = len >> 1;
    for (int t = tid; t < ROWS * 256; t += 256) {
      const int r   = t >> 8;
      const int bf  = t & 255;
      const int pos = bf & (half - 1);
      const int i0  = (r << 9) + ((bf & ~(half - 1)) << 1) + pos;
      const int i1  = i0 + half;
      const float ang = sign * 6.28318530717958647692f * (float)pos / (float)len;
      float s, c;
      __sincosf(ang, &s, &c);            // w = c + i s = e^{i*ang}
      const float xr = re[i1], xi = im[i1];
      const float tr = c * xr - s * xi;
      const float ti = c * xi + s * xr;
      const float yr = re[i0], yi = im[i0];
      re[i1] = yr - tr;  im[i1] = yi - ti;
      re[i0] = yr + tr;  im[i0] = yi + ti;
    }
    __syncthreads();
  }
}

// Per-row complex dot:  D_r = sum_c (h[r][c]*scale) * v[c]   (no conjugation,
// matching  h @ v  in the reference).  16 threads per row, LDS tree reduce.
__device__ __forceinline__ void row_dot(const float* __restrict__ hr_,
                                        const float* __restrict__ hi_,
                                        const float* __restrict__ vr,
                                        const float* __restrict__ vi,
                                        float* __restrict__ red,
                                        int tid, float scale,
                                        float& Dre, float& Dim) {
  const int r  = tid >> 4;
  const int ln = tid & 15;
  float dre = 0.f, dim = 0.f;
  for (int c = ln; c < HDIM; c += 16) {
    const float a = hr_[(r << 9) + c] * scale;
    const float b = hi_[(r << 9) + c] * scale;
    const float p = vr[c], q = vi[c];
    dre += a * p - b * q;
    dim += a * q + b * p;
  }
  red[tid]       = dre;
  red[256 + tid] = dim;
  __syncthreads();
  for (int s = 8; s > 0; s >>= 1) {
    if (ln < s) {
      red[tid]       += red[tid + s];
      red[256 + tid] += red[256 + tid + s];
    }
    __syncthreads();
  }
  Dre = red[(r << 4)];
  Dim = red[256 + (r << 4)];
  __syncthreads();   // allow red reuse
}

// ---------------------------------------------------------------------------
// Setup kernel 1: split W (fp32) into bf16 hi/lo pairs (re & im).
// ---------------------------------------------------------------------------
__global__ void urnn_cvt_w(const float* __restrict__ Wre,
                           const float* __restrict__ Wim,
                           unsigned short* __restrict__ Wrh_,
                           unsigned short* __restrict__ Wrl_,
                           unsigned short* __restrict__ Wih_,
                           unsigned short* __restrict__ Wil_) {
  const int idx = blockIdx.x * 256 + threadIdx.x;   // grid covers H*I exactly
  bf16_t* Wrh = (bf16_t*)Wrh_;
  bf16_t* Wrl = (bf16_t*)Wrl_;
  bf16_t* Wih = (bf16_t*)Wih_;
  bf16_t* Wil = (bf16_t*)Wil_;
  const float fr = Wre[idx];
  const bf16_t hr = (bf16_t)fr;
  Wrh[idx] = hr;
  Wrl[idx] = (bf16_t)(fr - (float)hr);
  const float fi = Wim[idx];
  const bf16_t hi = (bf16_t)fi;
  Wih[idx] = hi;
  Wil[idx] = (bf16_t)(fi - (float)hi);
}

// ---------------------------------------------------------------------------
// Setup kernel 2: trig tables for d1/d2/d3 and unit-normalized v1/v2.
// One block of 512 threads.
// ---------------------------------------------------------------------------
__global__ void urnn_setup(const float* __restrict__ ang,
                           const float* __restrict__ rre,
                           const float* __restrict__ rim,
                           float* __restrict__ par) {
  __shared__ float red[512];
  const int t = threadIdx.x;
  const float a1 = rre[t],       b1 = rim[t];
  const float a2 = rre[512 + t], b2 = rim[512 + t];

  red[t] = a1 * a1 + b1 * b1;
  __syncthreads();
  for (int s = 256; s > 0; s >>= 1) { if (t < s) red[t] += red[t + s]; __syncthreads(); }
  const float n1 = sqrtf(red[0]);
  __syncthreads();

  red[t] = a2 * a2 + b2 * b2;
  __syncthreads();
  for (int s = 256; s > 0; s >>= 1) { if (t < s) red[t] += red[t + s]; __syncthreads(); }
  const float n2 = sqrtf(red[0]);

  const float i1 = 1.f / (n1 + 1e-8f);
  const float i2 = 1.f / (n2 + 1e-8f);
  par[PV1R + t] = a1 * i1;  par[PV1I + t] = b1 * i1;
  par[PV2R + t] = a2 * i2;  par[PV2I + t] = b2 * i2;

  float s_, c_;
  __sincosf(ang[t],        &s_, &c_);  par[PD1C + t] = c_;  par[PD1S + t] = s_;
  __sincosf(ang[512 + t],  &s_, &c_);  par[PD2C + t] = c_;  par[PD2S + t] = s_;
  __sincosf(ang[1024 + t], &s_, &c_);  par[PD3C + t] = c_;  par[PD3S + t] = s_;
}

// ---------------------------------------------------------------------------
// Main fused kernel: 1024 blocks x 256 threads (8 wave32), 16 rows each.
//   Phase 0: async global->LDS stream of hx tile (overlaps the GEMM),
//            param staging, split-bf16 conversion of x rows.
//   Phase 1: WMMA GEMM  emb = x @ W^T (complex, split-bf16) -> LDS
//   Phase 2: d1 -> FFT -> HH(v1) -> perm/d2 -> IFFT -> HH(v2)/d3 -> +emb
//            -> ModReLU -> out
// ---------------------------------------------------------------------------
__global__ void __launch_bounds__(256)
urnn_main(const float* __restrict__ x,
          const float* __restrict__ hx_re,
          const float* __restrict__ hx_im,
          const float* __restrict__ beta,
          const int*   __restrict__ perm,
          const unsigned short* __restrict__ Wrh_,
          const unsigned short* __restrict__ Wrl_,
          const unsigned short* __restrict__ Wih_,
          const unsigned short* __restrict__ Wil_,
          const float* __restrict__ par,
          float* __restrict__ out) {
  extern __shared__ float smem[];
  float* h_re  = smem + L_HRE;
  float* h_im  = smem + L_HIM;
  float* t_re  = smem + L_TRE;
  float* t_im  = smem + L_TIM;
  float* e_re  = smem + L_ERE;
  float* e_im  = smem + L_EIM;
  float* red   = smem + L_RED;
  float* pbuf  = smem + L_PBUF;
  int*   lperm = (int*)(smem + L_PERM);
  // A staging (bf16 hi/lo) overlaps h_re region: consumed before h is written.
  bf16_t* a_hi = (bf16_t*)(smem + 0);      // 16x256 bf16 = 8 KB
  bf16_t* a_lo = (bf16_t*)(smem + 2048);   // 16x256 bf16 = 8 KB

  const bf16_t* Wrh = (const bf16_t*)Wrh_;
  const bf16_t* Wrl = (const bf16_t*)Wrl_;
  const bf16_t* Wih = (const bf16_t*)Wih_;
  const bf16_t* Wil = (const bf16_t*)Wil_;

  const int tid = threadIdx.x;
  const int b0  = blockIdx.x * ROWS;

#if URNN_ASYNC_LDS
  // ---- issue async DMA of the contiguous 16x512 hx tile into t_re/t_im.
  // These transfers stream from HBM while the WMMA GEMM below executes;
  // completion is enforced with s_wait_asynccnt + barrier before phase 2.
  {
    const float* gre = hx_re + (size_t)b0 * HDIM;
    const float* gim = hx_im + (size_t)b0 * HDIM;
    for (int idx = tid; idx < (ROWS * HDIM) / 4; idx += 256) {  // b128 chunks
      __builtin_amdgcn_global_load_async_to_lds_b128(
          (gv4i*)(gre + idx * 4), (lv4i*)(t_re + idx * 4), 0, 0);
      __builtin_amdgcn_global_load_async_to_lds_b128(
          (gv4i*)(gim + idx * 4), (lv4i*)(t_im + idx * 4), 0, 0);
    }
  }
#endif

  // ---- phase 0: stage params, perm, beta, and split x rows into bf16 hi/lo
  for (int c = tid; c < NPAR; c += 256) pbuf[c] = par[c];
  for (int c = tid; c < HDIM; c += 256) {
    pbuf[PBETA + c] = beta[c];
    lperm[c] = perm[c];
  }
  for (int idx = tid; idx < ROWS * IDIM; idx += 256) {
    const int r = idx >> 8, c = idx & 255;
    const float f = x[(b0 + r) * IDIM + c];
    const bf16_t h = (bf16_t)f;
    a_hi[idx] = h;
    a_lo[idx] = (bf16_t)(f - (float)h);
  }
  __syncthreads();

  // ---- phase 1: WMMA GEMM (split-bf16, fp32 accumulate)
  {
    const int lane = tid & 31;
    const int wave = tid >> 5;
    const int m  = lane & 15;
    const int kb = (lane >> 4) << 4;   // A K-chunk base: 0 or 16
    const int ks = (lane >> 4) << 4;   // B K-chunk base: 0 or 16
    const v8f z = {};
    v8f accR[4], accI[4];
#pragma unroll
    for (int j = 0; j < 4; ++j) { accR[j] = z; accI[j] = z; }

    for (int k0 = 0; k0 < IDIM; k0 += 32) {
      const v16bf ah = load_fragA(a_hi, m, k0, kb);
      const v16bf al = load_fragA(a_lo, m, k0, kb);
#pragma unroll
      for (int j = 0; j < 4; ++j) {
        const int h = ((wave << 2) + j) * 16 + m;   // column index (N)
        const v16bf wrh = load_fragB(Wrh, h, k0, ks);
        const v16bf wrl = load_fragB(Wrl, h, k0, ks);
        const v16bf wih = load_fragB(Wih, h, k0, ks);
        const v16bf wil = load_fragB(Wil, h, k0, ks);
        accR[j] = WMMA_BF16(ah, wrh, accR[j]);
        accR[j] = WMMA_BF16(ah, wrl, accR[j]);
        accR[j] = WMMA_BF16(al, wrh, accR[j]);
        accI[j] = WMMA_BF16(ah, wih, accI[j]);
        accI[j] = WMMA_BF16(ah, wil, accI[j]);
        accI[j] = WMMA_BF16(al, wih, accI[j]);
      }
      if (k0 + 32 < IDIM) {  // uniform; keep next k-step's W lines warm in L2/L0
        __builtin_prefetch(Wrh + ((wave << 2) * 16 + m) * IDIM + k0 + 32 + ks, 0, 0);
        __builtin_prefetch(Wih + ((wave << 2) * 16 + m) * IDIM + k0 + 32 + ks, 0, 0);
      }
    }

    // D writeback per documented 16x16 f32 C/D layout:
    //   VGPR v, lane L -> M = v + 8*(L>>4), N = L&15.
    const int mb = (lane >> 4) << 3;
#pragma unroll
    for (int j = 0; j < 4; ++j) {
      const int hc = ((wave << 2) + j) * 16 + (lane & 15);
#pragma unroll
      for (int v = 0; v < 8; ++v) {
        e_re[((mb + v) << 9) + hc] = accR[j][v];
        e_im[((mb + v) << 9) + hc] = accI[j][v];
      }
    }
  }

#if URNN_ASYNC_LDS
  __builtin_amdgcn_s_wait_asynccnt(0);   // this wave's DMA into t_re/t_im done
#endif
  __syncthreads();   // all waves' GEMM + DMA visible; A staging dead

  // ---- phase 2: h = d1 * (hx_re + i hx_im), stored bit-reversed
#if URNN_ASYNC_LDS
  for (int idx = tid; idx < ROWS * HDIM; idx += 256) {
    const int r = idx >> 9, c = idx & 511;
    const float xr = t_re[idx];          // staged by async DMA
    const float xi = t_im[idx];
    const float dc = pbuf[PD1C + c], dsn = pbuf[PD1S + c];
    const int cr = (int)(__brev((unsigned)c) >> 23);
    h_re[(r << 9) + cr] = dc * xr - dsn * xi;
    h_im[(r << 9) + cr] = dsn * xr + dc * xi;
  }
#else
  for (int idx = tid; idx < ROWS * HDIM; idx += 256) {
    const int r = idx >> 9, c = idx & 511;
    const float xr = hx_re[(b0 + r) * HDIM + c];
    const float xi = hx_im[(b0 + r) * HDIM + c];
    const float dc = pbuf[PD1C + c], dsn = pbuf[PD1S + c];
    const int cr = (int)(__brev((unsigned)c) >> 23);
    h_re[(r << 9) + cr] = dc * xr - dsn * xi;
    h_im[(r << 9) + cr] = dsn * xr + dc * xi;
  }
#endif
  __syncthreads();

  fft16x512(h_re, h_im, -1.f, tid);      // forward FFT

  // Householder 1: h -= 2 (h.v1) conj(v1)
  {
    float Dre, Dim;
    row_dot(h_re, h_im, pbuf + PV1R, pbuf + PV1I, red, tid, 1.f, Dre, Dim);
    const int r = tid >> 4, ln = tid & 15;
    for (int c = ln; c < HDIM; c += 16) {
      const float p = pbuf[PV1R + c], q = pbuf[PV1I + c];
      h_re[(r << 9) + c] -= 2.f * (Dre * p + Dim * q);
      h_im[(r << 9) + c] -= 2.f * (Dim * p - Dre * q);
    }
    __syncthreads();
  }

  // permute + d2, stored bit-reversed into t for IFFT (raw hx staging is dead)
  for (int idx = tid; idx < ROWS * HDIM; idx += 256) {
    const int r = idx >> 9, c = idx & 511;
    const int pc = lperm[c];
    const float hr = h_re[(r << 9) + pc];
    const float hi = h_im[(r << 9) + pc];
    const float dc = pbuf[PD2C + c], dsn = pbuf[PD2S + c];
    const int cr = (int)(__brev((unsigned)c) >> 23);
    t_re[(r << 9) + cr] = dc * hr - dsn * hi;
    t_im[(r << 9) + cr] = dsn * hr + dc * hi;
  }
  __syncthreads();

  fft16x512(t_re, t_im, +1.f, tid);      // inverse FFT (unnormalized; 1/N folded below)

  // Householder 2 (on h/N) + d3 + emb + ModReLU + store
  {
    float Dre, Dim;
    row_dot(t_re, t_im, pbuf + PV2R, pbuf + PV2I, red, tid, INV_N, Dre, Dim);
    const int r = tid >> 4, ln = tid & 15;
    float2* out2 = (float2*)out;
    for (int c = ln; c < HDIM; c += 16) {
      float hr = t_re[(r << 9) + c] * INV_N;
      float hi = t_im[(r << 9) + c] * INV_N;
      const float p = pbuf[PV2R + c], q = pbuf[PV2I + c];
      hr -= 2.f * (Dre * p + Dim * q);
      hi -= 2.f * (Dim * p - Dre * q);
      const float dc = pbuf[PD3C + c], dsn = pbuf[PD3S + c];
      const float zr = dc * hr - dsn * hi + e_re[(r << 9) + c];
      const float zi = dsn * hr + dc * hi + e_im[(r << 9) + c];
      const float mag = sqrtf(zr * zr + zi * zi);
      const float t1  = fmaxf(mag + pbuf[PBETA + c], 0.f);
      float outr, outi;
      if (mag > 0.f) { const float sc = t1 / mag; outr = sc * zr; outi = sc * zi; }
      else           { outr = t1; outi = 0.f; }   // angle(0)=0 -> exp(i0)=1
      out2[(size_t)(b0 + r) * HDIM + c] = make_float2(outr, outi);
    }
  }
}

// ---------------------------------------------------------------------------
// Host launcher
// ---------------------------------------------------------------------------
extern "C" void kernel_launch(void* const* d_in, const int* in_sizes, int n_in,
                              void* d_out, int out_size, void* d_ws, size_t ws_size,
                              hipStream_t stream) {
  (void)in_sizes; (void)n_in; (void)out_size; (void)ws_size;
  const float* x     = (const float*)d_in[0];
  const float* hx_re = (const float*)d_in[1];
  const float* hx_im = (const float*)d_in[2];
  const float* ang   = (const float*)d_in[3];
  const float* rre   = (const float*)d_in[4];
  const float* rim   = (const float*)d_in[5];
  const float* Wre   = (const float*)d_in[6];
  const float* Wim   = (const float*)d_in[7];
  const float* beta  = (const float*)d_in[8];
  const int*   perm  = (const int*)d_in[9];

  char* ws = (char*)d_ws;
  unsigned short* Wrh = (unsigned short*)(ws + 0);        // 512*256*2 B each
  unsigned short* Wrl = (unsigned short*)(ws + 262144);
  unsigned short* Wih = (unsigned short*)(ws + 524288);
  unsigned short* Wil = (unsigned short*)(ws + 786432);
  float* par = (float*)(ws + 1048576);                    // NPAR floats
  float* out = (float*)d_out;

  urnn_cvt_w<<<(HDIM * IDIM) / 256, 256, 0, stream>>>(Wre, Wim, Wrh, Wrl, Wih, Wil);
  urnn_setup<<<1, 512, 0, stream>>>(ang, rre, rim, par);
  urnn_main<<<B_ROWS / ROWS, 256, SMEM_BYTES, stream>>>(
      x, hx_re, hx_im, beta, perm, Wrh, Wrl, Wih, Wil, par, out);
}
